// AdvancedMoELayer_12403865550893
// MI455X (gfx1250) — matrix operand order repaired and checked by reading, not compiled, not run on hardware
//
#include <hip/hip_runtime.h>
#include <stdint.h>
#include <stddef.h>

// ---- problem constants (match reference) ----
constexpr int Bn = 1024;   // tokens
constexpr int Dn = 1024;   // input dim
constexpr int Hn = 2048;   // hidden dim
constexpr int On = 1024;   // output dim
constexpr int En = 8;      // experts
constexpr int CAP = Bn;    // worst-case tokens per expert
constexpr int MT  = 64;    // M tile per workgroup (4 x 16 per wave)
constexpr int NT  = 128;   // N tile per workgroup (8 waves x 16)

typedef __attribute__((ext_vector_type(16))) __bf16 v16bf;
typedef __attribute__((ext_vector_type(8)))  __bf16 v8bf;
typedef __attribute__((ext_vector_type(8)))  float  v8f;

// ------------------------------------------------------------------
// 0) zero the per-expert slot counters (ws is not re-poisoned per call)
// ------------------------------------------------------------------
__global__ void moe_zero_counts(int* __restrict__ cnt) {
    if (threadIdx.x < En) cnt[threadIdx.x] = 0;
}

// ------------------------------------------------------------------
// 1) router: logits -> softmax -> top-2 -> renormalize -> append to
//    per-expert token lists.  One thread per token.
// ------------------------------------------------------------------
__global__ void moe_router(const float* __restrict__ x,
                           const float* __restrict__ Wr,
                           const float* __restrict__ br,
                           int*   __restrict__ cnt,
                           int*   __restrict__ tokIdx,
                           int*   __restrict__ tokK,
                           float* __restrict__ tokW) {
    int t = blockIdx.x * blockDim.x + threadIdx.x;
    if (t >= Bn) return;

    float logit[En];
#pragma unroll
    for (int e = 0; e < En; ++e) logit[e] = br[e];

    const float* xr = x + (size_t)t * Dn;
    for (int d = 0; d < Dn; ++d) {
        float xv = xr[d];
        const float* wrow = Wr + (size_t)d * En;
#pragma unroll
        for (int e = 0; e < En; ++e) logit[e] += xv * wrow[e];
    }

    float mx = logit[0];
#pragma unroll
    for (int e = 1; e < En; ++e) mx = fmaxf(mx, logit[e]);
    float p[En]; float s = 0.f;
#pragma unroll
    for (int e = 0; e < En; ++e) { p[e] = __expf(logit[e] - mx); s += p[e]; }
    float inv_s = 1.f / s;
#pragma unroll
    for (int e = 0; e < En; ++e) p[e] *= inv_s;   // softmax probs

    // top-2 with lowest-index tie-break (strict >) like jax.lax.top_k
    int i0 = 0;
#pragma unroll
    for (int e = 1; e < En; ++e) if (p[e] > p[i0]) i0 = e;
    int i1 = (i0 == 0) ? 1 : 0;
#pragma unroll
    for (int e = 0; e < En; ++e) if (e != i0 && p[e] > p[i1]) i1 = e;

    float denom = p[i0] + p[i1] + 1e-6f;
    float w0 = p[i0] / denom;
    float w1 = p[i1] / denom;

    int s0 = atomicAdd(&cnt[i0], 1);
    tokIdx[i0 * CAP + s0] = t; tokK[i0 * CAP + s0] = 0; tokW[i0 * CAP + s0] = w0;
    int s1 = atomicAdd(&cnt[i1], 1);
    tokIdx[i1 * CAP + s1] = t; tokK[i1 * CAP + s1] = 1; tokW[i1 * CAP + s1] = w1;
}

// ------------------------------------------------------------------
// 2) gather selected token rows, fp32 -> bf16; zero-pad the partial
//    tail tile so GEMM tiles are always full.
// ------------------------------------------------------------------
__global__ void moe_gather_x(const float* __restrict__ x,
                             const int*   __restrict__ cnt,
                             const int*   __restrict__ tokIdx,
                             __bf16*      __restrict__ xg) {
    int e    = blockIdx.y;
    int slot = blockIdx.x;
    int c    = cnt[e];
    int cpad = (c + MT - 1) / MT * MT;
    if (slot >= cpad) return;

    __bf16* dst = xg + ((size_t)e * CAP + slot) * Dn;
    if (slot < c) {
        const float* src = x + (size_t)tokIdx[e * CAP + slot] * Dn;
        for (int i = threadIdx.x; i < Dn; i += blockDim.x)
            dst[i] = (__bf16)src[i];
    } else {
        for (int i = threadIdx.x; i < Dn; i += blockDim.x)
            dst[i] = (__bf16)0.f;
    }
}

// ------------------------------------------------------------------
// 3) expert GEMM on bf16 WMMA, software-pipelined (2-phase unroll).
//    A: bf16 activations in ws [E][CAP][KDIM], staged through a
//       double-buffered 64x32 LDS tile; all 8 ds_load_b128 issued
//       before the 4 v_wmma of a phase so DS waits retire incrementally.
//    W: fp32 weights [E][KDIM][NDIM], converted to bf16 in-register;
//       the next phase's A chunk and B fragment load while the current
//       phase's WMMAs execute (distinct even/odd register sets: no
//       pipeline register copies, compile-time LDS buffer parity).
//    FINAL: weighted scatter to yPart[k][token][O] (fp32).
// ------------------------------------------------------------------
template <int KDIM, int NDIM, bool RELU, bool FINAL>
__global__ __launch_bounds__(256)
void moe_expert_gemm(const __bf16* __restrict__ Abase,
                     const float*  __restrict__ Wbase,
                     const float*  __restrict__ biasBase,
                     __bf16*       __restrict__ Cbase,
                     const int*    __restrict__ cnt,
                     const int*    __restrict__ tokIdx,
                     const int*    __restrict__ tokK,
                     const float*  __restrict__ tokW,
                     float*        __restrict__ yPart) {
    static_assert(KDIM % 64 == 0, "two-phase pipeline needs KDIM % 64 == 0");
    const int e = blockIdx.z;
    const int c = cnt[e];
    const int mBase = blockIdx.y * MT;
    if (mBase >= c) return;                      // routed sparsity: skip empty tiles
    const int nBase = blockIdx.x * NT;

    const int tid  = threadIdx.x;
    const int lane = tid & 31;
    const int wave = tid >> 5;                   // 0..7
    const int l15  = lane & 15;
    const int hi   = lane >> 4;                  // 0 or 1
    const int n    = nBase + wave * 16 + l15;
    const int kbA  = hi * 8;                     // A frag: K in [kbA,kbA+8) U [kbA+16,kbA+24)
    const int kbB  = hi * 8;                     // B frag: K in [kbB,kbB+8) U [kbB+16,kbB+24)

    const __bf16* Aexp = Abase + (size_t)e * CAP * KDIM;
    const float*  Wexp = Wbase + (size_t)e * KDIM * NDIM;
    const float*  bexp = biasBase + (size_t)e * NDIM;

    // double-buffered A tile: 64 rows x 32 K of bf16 (2 x 4 KB)
    __shared__ __bf16 Alds[2][MT * 32];
    const int arow = tid >> 2;                   // 0..63
    const int aseg = (tid & 3) * 8;              // 16B segment within a 64B row
    const __bf16* Ag   = Aexp + (size_t)(mBase + arow) * KDIM + aseg;
    __bf16* const ldsw0 = &Alds[0][arow * 32 + aseg];
    __bf16* const ldsw1 = &Alds[1][arow * 32 + aseg];

    v8f acc[4];
#pragma unroll
    for (int i = 0; i < 4; ++i)
#pragma unroll
        for (int j = 0; j < 8; ++j) acc[i][j] = 0.f;

    auto loadB = [&](int k0) -> v16bf {
        v16bf r;
        const float* wcol = Wexp + (size_t)(k0 + kbB) * NDIM + n;
#pragma unroll
        for (int i = 0; i < 16; ++i)
            r[i] = (__bf16)wcol[(size_t)(i < 8 ? i : i + 8) * NDIM];
        return r;
    };

    // one K-phase: gather all 4 A fragments from LDS, then 4 WMMAs
    auto kphase = [&](int bufp, const v16bf& bfr) {
        v16bf af[4];
#pragma unroll
        for (int ms = 0; ms < 4; ++ms) {
            const __bf16* Ab = &Alds[bufp][(ms * 16 + l15) * 32 + kbA];
            v8bf alo = *(const v8bf*)(Ab);        // K = kbA .. kbA+7   (16B)
            v8bf ahi = *(const v8bf*)(Ab + 16);   // K = kbA+16 .. +23  (16B)
#pragma unroll
            for (int i = 0; i < 8; ++i) { af[ms][i] = alo[i]; af[ms][i + 8] = ahi[i]; }
        }
#pragma unroll
        for (int ms = 0; ms < 4; ++ms)
            acc[ms] = __builtin_amdgcn_wmma_f32_16x16x32_bf16(
                false, af[ms], false, bfr, (short)0, acc[ms], false, false);
    };

    // ---- prologue: stage K-step 0 into buffer 0 / bf_even ----
    *(v8bf*)(ldsw0) = *(const v8bf*)(Ag);
    v16bf bf_even = loadB(0);
    v16bf bf_odd;
    __syncthreads();

    for (int k0 = 0; k0 < KDIM; k0 += 64) {
        // ===== even phase: consume buffer 0 + bf_even, stage k0+32 =====
        {
            v8bf agn = *(const v8bf*)(Ag + k0 + 32);
            bf_odd = loadB(k0 + 32);
            kphase(0, bf_even);
            *(v8bf*)(ldsw1) = agn;
        }
        __syncthreads();

        // ===== odd phase: consume buffer 1 + bf_odd, stage k0+64 =====
        const bool more = (k0 + 64 < KDIM);
        if (more) {
            v8bf agn = *(const v8bf*)(Ag + k0 + 64);
            bf_even = loadB(k0 + 64);
            kphase(1, bf_odd);
            *(v8bf*)(ldsw0) = agn;
        } else {
            kphase(1, bf_odd);
        }
        __syncthreads();
    }

    // ---- epilogue: bias (+ReLU) then store / weighted scatter
    const float bias = bexp[n];
#pragma unroll
    for (int ms = 0; ms < 4; ++ms) {
#pragma unroll
        for (int r = 0; r < 8; ++r) {
            const int m = mBase + ms * 16 + r + hi * 8;   // C/D layout: VGPR r -> M = r + 8*hi
            float v = acc[ms][r] + bias;
            if (RELU) v = v > 0.f ? v : 0.f;
            if (!FINAL) {
                Cbase[((size_t)e * CAP + m) * NDIM + n] = (__bf16)v;
            } else {
                if (m < c) {
                    const int   t  = tokIdx[e * CAP + m];
                    const int   kk = tokK[e * CAP + m];
                    const float w  = tokW[e * CAP + m];
                    yPart[((size_t)kk * Bn + t) * On + n] = w * v;
                }
            }
        }
    }
}

// ------------------------------------------------------------------
// 4) fixed-order combine (deterministic): out = yPart[0] + yPart[1]
// ------------------------------------------------------------------
__global__ void moe_combine(const float* __restrict__ yPart,
                            float* __restrict__ out) {
    size_t i = (size_t)blockIdx.x * blockDim.x + threadIdx.x;
    out[i] = yPart[i] + yPart[(size_t)Bn * On + i];
}

// ------------------------------------------------------------------
extern "C" void kernel_launch(void* const* d_in, const int* in_sizes, int n_in,
                              void* d_out, int out_size, void* d_ws, size_t ws_size,
                              hipStream_t stream) {
    const float* x  = (const float*)d_in[0];
    const float* Wr = (const float*)d_in[1];
    const float* br = (const float*)d_in[2];
    const float* W1 = (const float*)d_in[3];
    const float* b1 = (const float*)d_in[4];
    const float* W2 = (const float*)d_in[5];
    const float* b2 = (const float*)d_in[6];
    const float* W3 = (const float*)d_in[7];
    const float* b3 = (const float*)d_in[8];
    float* out = (float*)d_out;

    // ---- workspace carve (~88.1 MB) ----
    char* p = (char*)d_ws;
    auto take = [&](size_t bytes) { char* q = p; p += (bytes + 255) & ~(size_t)255; return q; };
    int*    cnt    = (int*)   take(En * sizeof(int));
    int*    tokIdx = (int*)   take((size_t)En * CAP * sizeof(int));
    int*    tokK   = (int*)   take((size_t)En * CAP * sizeof(int));
    float*  tokW   = (float*) take((size_t)En * CAP * sizeof(float));
    __bf16* xg     = (__bf16*)take((size_t)En * CAP * Dn * sizeof(__bf16));
    __bf16* h1     = (__bf16*)take((size_t)En * CAP * Hn * sizeof(__bf16));
    __bf16* h2     = (__bf16*)take((size_t)En * CAP * Hn * sizeof(__bf16));
    float*  yPart  = (float*) take((size_t)2 * Bn * On * sizeof(float));

    moe_zero_counts<<<1, 32, 0, stream>>>(cnt);

    moe_router<<<Bn / 256, 256, 0, stream>>>(x, Wr, br, cnt, tokIdx, tokK, tokW);

    moe_gather_x<<<dim3(CAP, En), 256, 0, stream>>>(x, cnt, tokIdx, xg);

    // layer 1: [*,D] x [D,H] -> relu -> h1
    moe_expert_gemm<Dn, Hn, true, false>
        <<<dim3(Hn / NT, CAP / MT, En), 256, 0, stream>>>(
            xg, W1, b1, h1, cnt, nullptr, nullptr, nullptr, nullptr);

    // layer 2: [*,H] x [H,H] -> relu -> h2
    moe_expert_gemm<Hn, Hn, true, false>
        <<<dim3(Hn / NT, CAP / MT, En), 256, 0, stream>>>(
            h1, W2, b2, h2, cnt, nullptr, nullptr, nullptr, nullptr);

    // layer 3: [*,H] x [H,O] -> weighted scatter to yPart
    moe_expert_gemm<Hn, On, false, true>
        <<<dim3(On / NT, CAP / MT, En), 256, 0, stream>>>(
            h2, W3, b3, nullptr, cnt, tokIdx, tokK, tokW, yPart);

    moe_combine<<<(Bn * On) / 256, 256, 0, stream>>>(yPart, out);
}
